// QLSTM_65481071402622
// MI455X (gfx1250) — matrix operand order repaired and checked by reading, not compiled, not run on hardware
//
#include <hip/hip_runtime.h>
#include <hip/hip_bf16.h>

typedef __attribute__((ext_vector_type(8)))  __bf16 v8bf;
typedef __attribute__((ext_vector_type(16))) __bf16 v16bf;
typedef __attribute__((ext_vector_type(8)))  float  v8f;

#define T_STEPS 1024
#define BATCH   64
#define IN_DIM  256
#define HID     512
#define KDIM    768      // IN_DIM + HID
#define KPAD    776      // padded K stride (bf16 elems) to break LDS bank conflicts
#define NWG     16

__device__ __forceinline__ unsigned short f2bf(float f) {
  unsigned u = __builtin_bit_cast(unsigned, f);
  u += 0x7FFFu + ((u >> 16) & 1u);     // round-to-nearest-even bf16
  return (unsigned short)(u >> 16);
}
__device__ __forceinline__ float sigf(float x) {
  return 1.0f / (1.0f + __expf(-x));
}
__device__ __forceinline__ float tanhfast(float x) {
  return 2.0f / (1.0f + __expf(-2.0f * x)) - 1.0f;
}

// Persistent LSTM: 16 WGs x 256 threads (8 wave32). WG w owns hidden cols
// [w*32, w*32+32). Wave wv: m = wv>>1 (16-row tile), jt = wv&1 (16-col tile),
// and computes ALL FOUR gate tiles for those columns -> cell update is
// register-local; c-state lives in VGPRs for all 1024 steps.
__global__ void __launch_bounds__(256, 1)
lstm_persistent(const float* __restrict__ xin,   // (1024,64,256)
                const float* __restrict__ h0,    // (64,512)
                const float* __restrict__ c0,    // (64,512)
                const float* __restrict__ Wg,    // (768,2048) row-major
                const float* __restrict__ bg,    // (2048)
                float* __restrict__ out,         // outputs | hn | cn
                unsigned* __restrict__ bar)      // zeroed device barrier counter
{
  extern __shared__ unsigned short smem[];
  unsigned short* Wlds = smem;               // 128 cols x KPAD (col-major, bf16)
  unsigned short* Alds = smem + 128 * KPAD;  // 64 rows x KPAD (row-major, bf16)

  const int tid  = threadIdx.x;
  const int lane = tid & 31;
  const int wv   = tid >> 5;
  const int m    = wv >> 1;       // row-tile 0..3  (rows m*16..m*16+15)
  const int jt   = wv & 1;        // col-tile 0..1 within WG's 32 columns
  const int ln   = lane & 15;
  const int half = lane >> 4;
  const int w    = blockIdx.x;    // 0..15

  // ---- one-time: stage this WG's Wg slice (768 x 128) into LDS as bf16,
  // column-major with padded K stride. Local col c = q*32 + jj maps to
  // global gate column q*512 + w*32 + jj.
  {
    const int c    = tid >> 1;                    // 0..127
    const int kh   = tid & 1;                     // K half
    const int gcol = (c >> 5) * HID + w * 32 + (c & 31);
    unsigned short* dst = Wlds + c * KPAD;
    for (int k = kh * (KDIM / 2); k < kh * (KDIM / 2) + (KDIM / 2); ++k)
      dst[k] = f2bf(Wg[(size_t)k * (4 * HID) + gcol]);
  }

  const int jg = w * 32 + jt * 16 + ln;   // this lane's hidden column (0..511)
  const int r0 = m * 16 + half * 8;       // first batch-row of the acc block

  const float bF = bg[0 * HID + jg];
  const float bI = bg[1 * HID + jg];
  const float bG = bg[2 * HID + jg];
  const float bO = bg[3 * HID + jg];

  float c_st[8], h_st[8];
#pragma unroll
  for (int v = 0; v < 8; ++v) {
    c_st[v] = c0[(size_t)(r0 + v) * HID + jg];
    h_st[v] = 0.f;
  }

  __syncthreads();

  const size_t BH = (size_t)BATCH * HID;

  for (int t = 0; t < T_STEPS; ++t) {
    // ---- device-scope barrier: h(t-1) from every WG must be visible
    if (t > 0) {
      __threadfence();
      __syncthreads();
      if (tid == 0) {
        __hip_atomic_fetch_add(bar, 1u, __ATOMIC_RELEASE, __HIP_MEMORY_SCOPE_AGENT);
        const unsigned target = (unsigned)(NWG * t);
        while (__hip_atomic_load(bar, __ATOMIC_RELAXED, __HIP_MEMORY_SCOPE_AGENT) < target)
          __builtin_amdgcn_s_sleep(1);
        __builtin_amdgcn_fence(__ATOMIC_ACQUIRE, "agent");
      }
    }
    __syncthreads();

    // ---- stage A = [x_t | h_{t-1}] -> LDS bf16 (64 x 768, padded rows)
    {
      const float4* xs = (const float4*)(xin + (size_t)t * BATCH * IN_DIM);
      for (int e = tid; e < BATCH * IN_DIM / 4; e += 256) {
        float4 v = xs[e];
        const int row = e >> 6;            // 64 float4 per row
        const int cc  = (e & 63) * 4;
        unsigned lo = (unsigned)f2bf(v.x) | ((unsigned)f2bf(v.y) << 16);
        unsigned hi = (unsigned)f2bf(v.z) | ((unsigned)f2bf(v.w) << 16);
        *(uint2*)(Alds + row * KPAD + cc) = make_uint2(lo, hi);
      }
      const float4* hs = (const float4*)(t == 0 ? h0 : (out + (size_t)(t - 1) * BH));
      for (int e = tid; e < BATCH * HID / 4; e += 256) {
        float4 v = hs[e];
        const int row = e >> 7;            // 128 float4 per row
        const int cc  = IN_DIM + (e & 127) * 4;
        unsigned lo = (unsigned)f2bf(v.x) | ((unsigned)f2bf(v.y) << 16);
        unsigned hi = (unsigned)f2bf(v.z) | ((unsigned)f2bf(v.w) << 16);
        *(uint2*)(Alds + row * KPAD + cc) = make_uint2(lo, hi);
      }
    }
    __syncthreads();

    // ---- gate GEMM: four 16x16 f32 tiles (f,i,g,o), K = 768 in 24 steps
    v8f acc[4];
#pragma unroll
    for (int q = 0; q < 4; ++q)
#pragma unroll
      for (int v = 0; v < 8; ++v) acc[q][v] = 0.f;

    const unsigned short* arow = Alds + (m * 16 + ln) * KPAD;
#pragma unroll 4
    for (int kk = 0; kk < KDIM / 32; ++kk) {
      const int kb = kk * 32;
      // A fragment (16x32 bf16): lanes 0-15 K[kb..kb+8)+K[kb+16..kb+24),
      // lanes 16-31 shifted by 8 -> two 16B ds loads
      v8bf alo = *(const v8bf*)(arow + kb + half * 8);
      v8bf ahi = *(const v8bf*)(arow + kb + 16 + half * 8);
      v16bf afrag = __builtin_shufflevector(alo, ahi,
          0, 1, 2, 3, 4, 5, 6, 7, 8, 9, 10, 11, 12, 13, 14, 15);
#pragma unroll
      for (int q = 0; q < 4; ++q) {
        // B fragment (32x16 bf16): lane = column, K[kb+8h..+8) + K[kb+16+8h..+8)
        const unsigned short* bcol =
            Wlds + (q * 32 + jt * 16 + ln) * KPAD + kb + half * 8;
        v8bf blo = *(const v8bf*)(bcol);
        v8bf bhi = *(const v8bf*)(bcol + 16);
        v16bf bfrag = __builtin_shufflevector(blo, bhi,
            0, 1, 2, 3, 4, 5, 6, 7, 8, 9, 10, 11, 12, 13, 14, 15);
        acc[q] = __builtin_amdgcn_wmma_f32_16x16x32_bf16(
            false, afrag, false, bfrag, (short)0, acc[q], false, false);
      }
    }

    // ---- LSTM cell update (register-local) + stream h_t to outputs[t]
    float* hdst = out + (size_t)t * BH;
#pragma unroll
    for (int v = 0; v < 8; ++v) {
      float f  = sigf(acc[0][v] + bF);
      float i  = sigf(acc[1][v] + bI);
      float g  = tanhfast(acc[2][v] + bG);
      float o  = sigf(acc[3][v] + bO);
      float cn = f * c_st[v] + i * g;
      float hn = o * tanhfast(cn);
      c_st[v] = cn;
      h_st[v] = hn;
      hdst[(size_t)(r0 + v) * HID + jg] = hn;
    }
  }

  // ---- tail outputs: hn, cn
  float* hn_out = out + (size_t)T_STEPS * BH;
  float* cn_out = hn_out + BH;
#pragma unroll
  for (int v = 0; v < 8; ++v) {
    hn_out[(size_t)(r0 + v) * HID + jg] = h_st[v];
    cn_out[(size_t)(r0 + v) * HID + jg] = c_st[v];
  }
}

extern "C" void kernel_launch(void* const* d_in, const int* in_sizes, int n_in,
                              void* d_out, int out_size, void* d_ws, size_t ws_size,
                              hipStream_t stream) {
  const float* xin = (const float*)d_in[0];
  const float* h0  = (const float*)d_in[1];
  const float* c0  = (const float*)d_in[2];
  const float* Wg  = (const float*)d_in[3];
  const float* bg  = (const float*)d_in[4];
  // d_in[5..12] (We,be,Wd,bd,Wqkv,bqkv,Wo,bo) are dead code in the reference.
  float*    out = (float*)d_out;
  unsigned* bar = (unsigned*)d_ws;

  const size_t shmem = (size_t)(128 + 64) * KPAD * sizeof(unsigned short); // ~291 KB
  hipFuncSetAttribute((const void*)lstm_persistent,
                      hipFuncAttributeMaxDynamicSharedMemorySize, (int)shmem);

  hipMemsetAsync(d_ws, 0, 256, stream);  // barrier counter = 0 (capture-safe)
  lstm_persistent<<<dim3(NWG), dim3(256), shmem, stream>>>(xin, h0, c0, Wg, bg, out, bar);
}
